// MorphLayer_31997506355599
// MI455X (gfx1250) — matrix operand order repaired and checked by reading, not compiled, not run on hardware
//
#include <hip/hip_runtime.h>
#include <hip/hip_bf16.h>
#include <math.h>

typedef __attribute__((ext_vector_type(16))) __bf16 v16bf;
typedef __attribute__((ext_vector_type(8)))  __bf16 v8bf;
typedef __attribute__((ext_vector_type(8)))  float  v8f;
typedef __attribute__((ext_vector_type(4)))  float  v4f;

#define BATCH   65536
#define DMODEL  1024
#define RANK    64

__device__ inline v8f wmma_bf16(v16bf a, v16bf b, v8f c) {
  // D = A(16x32 bf16) * B(32x16 bf16) + C(16x16 f32)
  return __builtin_amdgcn_wmma_f32_16x16x32_bf16(
      /*neg_a=*/false, a, /*neg_b=*/false, b,
      /*c_mod=*/(short)0, c, /*reuse_a=*/false, /*reuse_b=*/false);
}

// ---------------------------------------------------------------------------
// Kernel 0: zero accumulators; repack weights to bf16 in WMMA B-layout.
// B-layout (32x16 bf16 chunk): entry ((kb*4+t)*32+lane)*16 + j holds
//   W[kb*32 + (lane>>4)*16 + j][t*16 + (lane&15)]
// ---------------------------------------------------------------------------
__global__ void morph_prep(const float* __restrict__ dw,
                           const float* __restrict__ kw,
                           const float* __restrict__ vw,
                           __bf16* __restrict__ Wb,
                           __bf16* __restrict__ Kb,
                           __bf16* __restrict__ Vb,
                           float* __restrict__ xsum,
                           float* __restrict__ delta_g) {
  const int t = threadIdx.x;  // 256 threads, 1 block
  for (int i = t; i < DMODEL; i += 256) xsum[i] = 0.0f;
  for (int i = t; i < RANK * RANK; i += 256) delta_g[i] = 0.0f;

  // down_w: 32 K-chunks x 4 N-tiles x 32 lanes
  for (int e = t; e < 32 * 4 * 32; e += 256) {
    const int l  = e & 31;
    const int tt = (e >> 5) & 3;
    const int kb = e >> 7;
    const int col   = tt * 16 + (l & 15);
    const int rbase = kb * 32 + (l >> 4) * 16;
    __bf16* dst = Wb + e * 16;
#pragma unroll
    for (int j = 0; j < 16; ++j)
      dst[j] = (__bf16)dw[(rbase + j) * RANK + col];
  }
  // key_w / val_w: 2 K-chunks x 4 N-tiles x 32 lanes
  for (int e = t; e < 2 * 4 * 32; e += 256) {
    const int l  = e & 31;
    const int tt = (e >> 5) & 3;
    const int kb = e >> 7;
    const int col   = tt * 16 + (l & 15);
    const int rbase = kb * 32 + (l >> 4) * 16;
    __bf16* dk = Kb + e * 16;
    __bf16* dv = Vb + e * 16;
#pragma unroll
    for (int j = 0; j < 16; ++j) {
      dk[j] = (__bf16)kw[(rbase + j) * RANK + col];
      dv[j] = (__bf16)vw[(rbase + j) * RANK + col];
    }
  }
}

// ---------------------------------------------------------------------------
// Kernel 1: column sums of x (for x_mean). Block b handles 256 rows; lanes
// read consecutive float4 columns -> fully coalesced; flush via f32 atomics.
// ---------------------------------------------------------------------------
__global__ __launch_bounds__(256) void morph_colsum(const float* __restrict__ x,
                                                    float* __restrict__ xsum) {
  const int t = threadIdx.x;
  const size_t rowbase = (size_t)blockIdx.x * 256;
  float a0 = 0.f, a1 = 0.f, a2 = 0.f, a3 = 0.f;
  const float* p = x + rowbase * DMODEL + t * 4;
  for (int r = 0; r < 256; ++r) {
    v4f v = *(const v4f*)(p + (size_t)r * DMODEL);
    a0 += v[0]; a1 += v[1]; a2 += v[2]; a3 += v[3];
  }
  unsafeAtomicAdd(&xsum[t * 4 + 0], a0);
  unsafeAtomicAdd(&xsum[t * 4 + 1], a1);
  unsafeAtomicAdd(&xsum[t * 4 + 2], a2);
  unsafeAtomicAdd(&xsum[t * 4 + 3], a3);
}

// ---------------------------------------------------------------------------
// Kernel 2: fused streaming pass. 4 waves/block, each wave owns 16-row tiles:
//   h = gelu(x_tile @ down_w + b)   (32x v_wmma_f32_16x16x32_bf16, K=1024)
//   k = h @ key_w + b ; v = h @ val_w + b
//   delta += v^T k                  (16 WMMAs, K zero-padded 16->32)
// ---------------------------------------------------------------------------
__global__ __launch_bounds__(128) void morph_main(
    const float* __restrict__ x,
    const float* __restrict__ down_b,
    const float* __restrict__ key_b,
    const float* __restrict__ val_b,
    const __bf16* __restrict__ Wb,
    const __bf16* __restrict__ Kb,
    const __bf16* __restrict__ Vb,
    float* __restrict__ delta_g) {
  __shared__ __align__(32) __bf16 h_lds[4][16 * 64];   // row-major h per wave
  __shared__ __align__(32) __bf16 kT[4][64 * 32];      // k^T, K padded to 32
  __shared__ __align__(32) __bf16 vT[4][64 * 32];      // v^T, K padded to 32
  __shared__ float delta_lds[RANK * RANK];

  const int tid  = threadIdx.x;
  const int wave = tid >> 5;
  const int lane = tid & 31;
  const int ln16 = lane & 15;
  const int hi   = lane >> 4;  // 0 or 1 (lane half)

  for (int i = tid; i < RANK * RANK; i += 128) delta_lds[i] = 0.0f;
  {
    v8bf z;
#pragma unroll
    for (int j = 0; j < 8; ++j) z[j] = (__bf16)0.0f;
    for (int r = lane; r < 64; r += 32) {   // zero-pad K=16..31 once
      *(v8bf*)&kT[wave][r * 32 + 16] = z;
      *(v8bf*)&kT[wave][r * 32 + 24] = z;
      *(v8bf*)&vT[wave][r * 32 + 16] = z;
      *(v8bf*)&vT[wave][r * 32 + 24] = z;
    }
  }
  __syncthreads();

  v8f dc[4][4];  // delta accumulators, kept in VGPRs across tiles
#pragma unroll
  for (int a = 0; a < 4; ++a)
#pragma unroll
    for (int b = 0; b < 4; ++b)
#pragma unroll
      for (int j = 0; j < 8; ++j) dc[a][b][j] = 0.0f;

  const int wave_slot = blockIdx.x * 4 + wave;  // 0..1023
  for (int tix = 0; tix < 4; ++tix) {
    const int tile = wave_slot * 4 + tix;       // 0..4095
    const size_t row = (size_t)tile * 16 + ln16;
    const float* xrow = x + row * DMODEL;
    __builtin_prefetch(xrow + DMODEL * 16, 0, 0);  // warm next tile

    // ---- h = x_tile @ down_w (f32 accum) ----
    v8f hc[4];
#pragma unroll
    for (int t = 0; t < 4; ++t)
#pragma unroll
      for (int j = 0; j < 8; ++j) hc[t][j] = 0.0f;

    for (int kb = 0; kb < 32; ++kb) {
      const int kbase = kb * 32 + hi * 16;  // A-layout: lanes 16-31 offset +16
      v4f f0 = *(const v4f*)(xrow + kbase);
      v4f f1 = *(const v4f*)(xrow + kbase + 4);
      v4f f2 = *(const v4f*)(xrow + kbase + 8);
      v4f f3 = *(const v4f*)(xrow + kbase + 12);
      v16bf a;
#pragma unroll
      for (int j = 0; j < 4; ++j) {
        a[j]      = (__bf16)f0[j];
        a[4 + j]  = (__bf16)f1[j];
        a[8 + j]  = (__bf16)f2[j];
        a[12 + j] = (__bf16)f3[j];
      }
      const __bf16* wb = Wb + (size_t)kb * 2048 + lane * 16;
#pragma unroll
      for (int t = 0; t < 4; ++t) {
        v16bf b = *(const v16bf*)(wb + t * 512);
        hc[t] = wmma_bf16(a, b, hc[t]);
      }
    }

    // ---- bias + exact-erf GELU; stage h row-major in LDS ----
#pragma unroll
    for (int t = 0; t < 4; ++t) {
      const float bia = down_b[t * 16 + ln16];
#pragma unroll
      for (int j = 0; j < 8; ++j) {
        float z = hc[t][j] + bia;
        float g = 0.5f * z * (1.0f + erff(z * 0.70710678118654752f));
        h_lds[wave][(j + hi * 8) * 64 + t * 16 + ln16] = (__bf16)g;
      }
    }
    asm volatile("s_wait_dscnt 0x0" ::: "memory");

    // ---- k = h @ key_w, v = h @ val_w ----
    v8f kc[4], vc[4];
#pragma unroll
    for (int t = 0; t < 4; ++t)
#pragma unroll
      for (int j = 0; j < 8; ++j) { kc[t][j] = 0.0f; vc[t][j] = 0.0f; }

#pragma unroll
    for (int kh = 0; kh < 2; ++kh) {
      const __bf16* hp = &h_lds[wave][ln16 * 64 + kh * 32 + hi * 16];
      v8bf lo  = *(const v8bf*)hp;
      v8bf hi8 = *(const v8bf*)(hp + 8);
      v16bf ha;
#pragma unroll
      for (int j = 0; j < 8; ++j) { ha[j] = lo[j]; ha[8 + j] = hi8[j]; }
      const __bf16* kbp = Kb + kh * 2048 + lane * 16;
      const __bf16* vbp = Vb + kh * 2048 + lane * 16;
#pragma unroll
      for (int t = 0; t < 4; ++t) {
        kc[t] = wmma_bf16(ha, *(const v16bf*)(kbp + t * 512), kc[t]);
        vc[t] = wmma_bf16(ha, *(const v16bf*)(vbp + t * 512), vc[t]);
      }
    }

    // ---- biases; stage k^T, v^T (A/B layouts for the outer-product WMMA) ----
#pragma unroll
    for (int t = 0; t < 4; ++t) {
      const float kbia = key_b[t * 16 + ln16];
      const float vbia = val_b[t * 16 + ln16];
      v8bf kp, vp;
#pragma unroll
      for (int j = 0; j < 8; ++j) {
        kp[j] = (__bf16)(kc[t][j] + kbia);
        vp[j] = (__bf16)(vc[t][j] + vbia);
      }
      const int s = t * 16 + ln16;  // rank index (column of D-layout)
      *(v8bf*)&kT[wave][s * 32 + hi * 8] = kp;  // rows m = hi*8..hi*8+7
      *(v8bf*)&vT[wave][s * 32 + hi * 8] = vp;
    }
    asm volatile("s_wait_dscnt 0x0" ::: "memory");

    // ---- delta[r][s] += sum_m v[m][r] * k[m][s]  (K=16 padded to 32) ----
#pragma unroll
    for (int rt = 0; rt < 4; ++rt) {
      const __bf16* vp = &vT[wave][(rt * 16 + ln16) * 32 + hi * 16];
      v8bf l0 = *(const v8bf*)vp;         // m = hi*16..+7 (valid for hi=0, zeros for hi=1)
      v8bf l1 = *(const v8bf*)(vp + 8);   // m = hi*16+8..+15 (valid for hi=0, zeros for hi=1)
      v16bf av;
#pragma unroll
      for (int j = 0; j < 8; ++j) { av[j] = l0[j]; av[8 + j] = l1[j]; }
#pragma unroll
      for (int st = 0; st < 4; ++st) {
        v16bf bk = *(const v16bf*)&kT[wave][(st * 16 + ln16) * 32 + hi * 16];
        dc[rt][st] = wmma_bf16(av, bk, dc[rt][st]);
      }
    }
  }

  // flush per-wave delta into LDS, then block flush to global with atomics
#pragma unroll
  for (int rt = 0; rt < 4; ++rt)
#pragma unroll
    for (int st = 0; st < 4; ++st)
#pragma unroll
      for (int j = 0; j < 8; ++j)
        atomicAdd(&delta_lds[(rt * 16 + j + hi * 8) * RANK + st * 16 + ln16],
                  dc[rt][st][j]);
  __syncthreads();
  for (int i = tid; i < RANK * RANK; i += 128)
    unsafeAtomicAdd(&delta_g[i], delta_lds[i]);
}

// ---------------------------------------------------------------------------
// Kernel 3: finalize. cosine/decay, M_new, one-sided (Hestenes) Jacobi SVD,
// context = sum of 10 largest-norm rotated columns (= sigma_i * u_i).
// ---------------------------------------------------------------------------
__global__ __launch_bounds__(64) void morph_finalize(
    const float* __restrict__ xsum,
    const float* __restrict__ delta_g,
    const float* __restrict__ M,
    const float* __restrict__ centroid,
    const float* __restrict__ n_tokens,
    float* __restrict__ out) {
  __shared__ float G[64 * 65];  // column-major, stride 65 (bank-conflict free)
  __shared__ float red[3][64];
  __shared__ float norms[64];
  __shared__ int   topi[10];
  const int t = threadIdx.x;

  // cosine similarity terms
  float d0 = 0.f, d1 = 0.f, d2 = 0.f;
  for (int i = t; i < DMODEL; i += 64) {
    float xm = xsum[i] * (1.0f / (float)BATCH);
    float c  = centroid[i];
    d0 += xm * c; d1 += xm * xm; d2 += c * c;
  }
  red[0][t] = d0; red[1][t] = d1; red[2][t] = d2;
  __syncthreads();
  for (int off = 32; off > 0; off >>= 1) {
    if (t < off) {
      red[0][t] += red[0][t + off];
      red[1][t] += red[1][t + off];
      red[2][t] += red[2][t + off];
    }
    __syncthreads();
  }
  const float dot = red[0][0];
  const float n1 = sqrtf(red[1][0]), n2 = sqrtf(red[2][0]);
  const float sim = dot / (fmaxf(n1, 1e-8f) * fmaxf(n2, 1e-8f));
  float decay = powf(0.95f, 2.0f - sim);
  decay = fminf(fmaxf(decay, 0.5f), 0.99f);
  if (n_tokens[0] < 1.0f) decay = 0.95f;
  __syncthreads();

  // M_new -> G (column-major), track max|M_new|
  float mx = 0.0f;
  for (int e = t; e < RANK * RANK; e += 64) {
    const int r = e >> 6, c = e & 63;
    float m = decay * M[e] + 0.1f * (delta_g[e] * (1.0f / (float)BATCH));
    G[c * 65 + r] = m;
    mx = fmaxf(mx, fabsf(m));
  }
  red[0][t] = mx;
  __syncthreads();
  for (int off = 32; off > 0; off >>= 1) {
    if (t < off) red[0][t] = fmaxf(red[0][t], red[0][t + off]);
    __syncthreads();
  }
  const float maxabs = red[0][0];

  // one-sided Jacobi: tournament rounds, 32 independent pairs per round
  for (int sweep = 0; sweep < 12; ++sweep) {
    for (int r = 0; r < 63; ++r) {
      __syncthreads();
      if (t < 32) {
        const int p = (t == 0) ? 0 : 1 + ((t - 1 + r) % 63);
        const int q = 1 + ((62 - t + r) % 63);
        float app = 0.f, aqq = 0.f, apq = 0.f;
        for (int i = 0; i < 64; ++i) {
          float a = G[p * 65 + i], b = G[q * 65 + i];
          app += a * a; aqq += b * b; apq += a * b;
        }
        if (fabsf(apq) > 1e-18f) {
          float zeta = (aqq - app) / (2.0f * apq);
          float tt = copysignf(1.0f, zeta) /
                     (fabsf(zeta) + sqrtf(1.0f + zeta * zeta));
          float cs = 1.0f / sqrtf(1.0f + tt * tt);
          float sn = cs * tt;
          for (int i = 0; i < 64; ++i) {
            float a = G[p * 65 + i], b = G[q * 65 + i];
            G[p * 65 + i] = cs * a - sn * b;
            G[q * 65 + i] = sn * a + cs * b;
          }
        }
      }
    }
  }
  __syncthreads();

  // column norms; select top-10; context = sum of those columns
  {
    float s = 0.f;
    for (int i = 0; i < 64; ++i) { float a = G[t * 65 + i]; s += a * a; }
    norms[t] = s;
  }
  __syncthreads();
  if (t == 0) {
    float tmp[64];
    for (int i = 0; i < 64; ++i) tmp[i] = norms[i];
    for (int k = 0; k < 10; ++k) {
      int bi = 0; float bv = tmp[0];
      for (int i = 1; i < 64; ++i)
        if (tmp[i] > bv) { bv = tmp[i]; bi = i; }
      topi[k] = bi;
      tmp[bi] = -1.0f;
    }
  }
  __syncthreads();
  float ctx = 0.0f;
  for (int k = 0; k < 10; ++k) ctx += G[topi[k] * 65 + t];
  if (maxabs < 1e-6f) ctx = 0.0f;
  out[t] = ctx;
}

// ---------------------------------------------------------------------------
extern "C" void kernel_launch(void* const* d_in, const int* in_sizes, int n_in,
                              void* d_out, int out_size, void* d_ws, size_t ws_size,
                              hipStream_t stream) {
  const float* x        = (const float*)d_in[0];
  const float* down_w   = (const float*)d_in[1];
  const float* down_b   = (const float*)d_in[2];
  const float* key_w    = (const float*)d_in[3];
  const float* key_b    = (const float*)d_in[4];
  const float* val_w    = (const float*)d_in[5];
  const float* val_b    = (const float*)d_in[6];
  const float* M        = (const float*)d_in[7];
  const float* centroid = (const float*)d_in[8];
  const float* n_tokens = (const float*)d_in[9];
  float* out = (float*)d_out;

  char* ws = (char*)d_ws;
  __bf16* Wb      = (__bf16*)(ws);            // 131072 B: down_w bf16 B-layout
  __bf16* Kb      = (__bf16*)(ws + 131072);   //   8192 B: key_w
  __bf16* Vb      = (__bf16*)(ws + 139264);   //   8192 B: val_w
  float*  xsum    = (float*)(ws + 147456);    //   4096 B: column sums of x
  float*  delta_g = (float*)(ws + 151552);    //  16384 B: sum_b v k^T

  morph_prep<<<1, 256, 0, stream>>>(down_w, key_w, val_w, Wb, Kb, Vb, xsum, delta_g);
  morph_colsum<<<BATCH / 256, 256, 0, stream>>>(x, xsum);
  morph_main<<<256, 128, 0, stream>>>(x, down_b, key_b, val_b, Wb, Kb, Vb, delta_g);
  morph_finalize<<<1, 64, 0, stream>>>(xsum, delta_g, M, centroid, n_tokens, out);
}
